// DirectAttention_41180146434121
// MI455X (gfx1250) — matrix operand order repaired and checked by reading, not compiled, not run on hardware
//
#include <hip/hip_runtime.h>
#include <hip/hip_bf16.h>

// Problem constants (match reference)
#define BB 4
#define NN 1024
#define CCH 512
#define HHD 8
#define HD 64
#define BN (BB * NN)
#define C3 (3 * CCH)

// Flip to 0 if the assembler rejects the async-to-LDS mnemonic.
#define USE_ASYNC_LDS 1

typedef unsigned short u16;
typedef unsigned int u32;
typedef unsigned long long u64;
typedef __attribute__((ext_vector_type(16))) __bf16 v16bf;
typedef __attribute__((ext_vector_type(8)))  float  v8f;

union Frag16 { v16bf v; u32 u[8]; u16 s[16]; };

__device__ __forceinline__ u16 f2bf(float f) {
  u32 u = __float_as_uint(f);
  u32 r = u + 0x7FFFu + ((u >> 16) & 1u);   // round-to-nearest-even
  return (u16)(r >> 16);
}

__device__ __forceinline__ v8f wmma_bf16(Frag16 a, Frag16 b, v8f c) {
  return __builtin_amdgcn_wmma_f32_16x16x32_bf16(
      /*neg_a=*/false, a.v, /*neg_b=*/false, b.v,
      /*c_mod=*/(short)0, c, /*reuse_a=*/false, /*reuse_b=*/false);
}

// A-fragment K offset for lane-half hf, dword i (16-bit A 16x32 layout)
__device__ __forceinline__ int a_koff(int i, int hf) {
  return ((i >> 2) << 4) + hf * 8 + ((i & 3) << 1);
}
// A fragment from a row-major row (K pairs contiguous)
__device__ __forceinline__ void loadA(Frag16& f, const u16* row, int hf) {
#pragma unroll
  for (int i = 0; i < 8; ++i) f.u[i] = *(const u32*)(row + a_koff(i, hf));
}
// B fragment (16-bit B 32x16 layout, same K order as A fragment)
// base points at this lane's K-major column.
__device__ __forceinline__ void loadB(Frag16& f, const u16* base, int hf) {
#pragma unroll
  for (int i = 0; i < 8; ++i) f.u[i] = *(const u32*)(base + a_koff(i, hf));
}

// Stage a 16KB tile (16 rows x 512 bf16) global -> LDS, 256 threads.
__device__ __forceinline__ void stage_tile(u16* dstLds, const u16* src, int tid) {
#if USE_ASYNC_LDS
  // CDNA5 async copy: per-lane B128 global->LDS, tracked by ASYNCcnt.
#pragma unroll
  for (int p = 0; p < 4; ++p) {
    u32 ldsoff = (u32)(size_t)dstLds + (u32)(tid * 16 + p * 4096);
    u64 gaddr  = (u64)(size_t)src + (u64)(tid * 16 + p * 4096);
    asm volatile("global_load_async_to_lds_b128 %0, %1, off"
                 :: "v"(ldsoff), "v"(gaddr) : "memory");
  }
  asm volatile("s_wait_asynccnt 0" ::: "memory");
#else
  const u32* s = (const u32*)src;
  u32* d = (u32*)dstLds;
#pragma unroll
  for (int i = 0; i < 16; ++i) d[tid + 256 * i] = s[tid + 256 * i];
#endif
}

// ---------------------------------------------------------------- weights->bf16
__global__ void cvt_weights(const float* __restrict__ qkv_w,
                            const float* __restrict__ out_w,
                            u16* __restrict__ wqkv, u16* __restrict__ wout) {
  int i = blockIdx.x * blockDim.x + threadIdx.x;
  if (i < C3 * CCH) wqkv[i] = f2bf(qkv_w[i]);
  if (i < CCH * CCH) wout[i] = f2bf(out_w[i]);
}

// ---------------------------------------------------------------- LayerNorm -> bf16
__global__ __launch_bounds__(256) void ln_kernel(const float* __restrict__ x,
                                                 const float* __restrict__ gamma,
                                                 const float* __restrict__ beta,
                                                 u16* __restrict__ xn) {
  int wave = threadIdx.x >> 5, lane = threadIdx.x & 31;
  int row = blockIdx.x * 8 + wave;          // row in [0, BN)
  const float* xr = x + (size_t)row * CCH;
  float vals[16];
  float s = 0.f, s2 = 0.f;
#pragma unroll
  for (int t = 0; t < 16; ++t) {
    float v = xr[t * 32 + lane];
    vals[t] = v; s += v; s2 += v * v;
  }
#pragma unroll
  for (int m = 16; m > 0; m >>= 1) { s += __shfl_xor(s, m, 32); s2 += __shfl_xor(s2, m, 32); }
  float mu = s * (1.0f / CCH);
  float var = s2 * (1.0f / CCH) - mu * mu;
  float inv = rsqrtf(var + 1e-5f);
#pragma unroll
  for (int t = 0; t < 16; ++t) {
    int c = t * 32 + lane;
    xn[(size_t)row * CCH + c] = f2bf((vals[t] - mu) * inv * gamma[c] + beta[c]);
  }
}

// ---------------------------------------------------------------- QKV GEMM (WMMA bf16)
// xn [BN,C] @ qkv_w^T -> scatter q[bh,n,d], k[bh,n,d], vT[bh,d,n] (bf16)
__global__ __launch_bounds__(256) void qkv_gemm(const u16* __restrict__ xn,
                                                const u16* __restrict__ wqkv,
                                                const float* __restrict__ qkv_b,
                                                u16* __restrict__ qb,
                                                u16* __restrict__ kb,
                                                u16* __restrict__ vT) {
  __shared__ u16 As[16 * CCH];              // 16 KB A tile
  int rowBase = blockIdx.x * 16;
  int tid = threadIdx.x;
  stage_tile(As, xn + (size_t)rowBase * CCH, tid);
  __syncthreads();
  int wave = tid >> 5, lane = tid & 31;
  int m = lane & 15, hf = lane >> 4;

  for (int it = 0; it < 12; ++it) {
    int ct = wave + it * 8;                 // col tile 0..95
    int oc = ct * 16 + m;                   // output column for this lane
    const u16* wbase = wqkv + (size_t)oc * CCH;
    v8f acc = {};
    Frag16 bw0, bw1;
    loadB(bw0, wbase, hf);                  // prefetch chunk 0
#pragma unroll
    for (int kc = 0; kc < 16; ++kc) {
      Frag16& cur = (kc & 1) ? bw1 : bw0;
      Frag16& nxt = (kc & 1) ? bw0 : bw1;
      if (kc + 1 < 16) loadB(nxt, wbase + (kc + 1) * 32, hf);  // pipeline B
      Frag16 a;
      loadA(a, &As[m * CCH + kc * 32], hf);
      acc = wmma_bf16(a, cur, acc);
    }
    float bias = qkv_b[oc];
    int sel = oc / CCH, rem = oc % CCH;
    int h = rem / HD, d = rem % HD;
#pragma unroll
    for (int j = 0; j < 8; ++j) {
      int row = rowBase + j + hf * 8;       // row in [0, BN)
      int b = row >> 10, n = row & (NN - 1);
      int bh = b * HHD + h;
      u16 bv = f2bf(acc[j] + bias);
      if (sel == 0)      qb[((size_t)bh * NN + n) * HD + d] = bv;
      else if (sel == 1) kb[((size_t)bh * NN + n) * HD + d] = bv;
      else               vT[((size_t)bh * HD + d) * NN + n] = bv;
    }
  }
}

// ---------------------------------------------------------------- fused attention
// per block: 16 query rows, full 1024 keys. scores in LDS, radix-select top-k,
// softmax in-place (bf16 attn aliases score rows), then attn@V via WMMA with
// an 8-wave K-split reduction.
__global__ __launch_bounds__(256) void attn_kernel(const u16* __restrict__ qb,
                                                   const u16* __restrict__ kb,
                                                   const u16* __restrict__ vT,
                                                   const float* __restrict__ attn_bias,
                                                   const float* __restrict__ adj,
                                                   const float* __restrict__ adj_scale,
                                                   const float* __restrict__ sth,
                                                   u16* __restrict__ attended) {
  __shared__ float Sc[16 * NN];             // 64 KB score tile
  __shared__ float Red[4 * 256];            // 4 KB attend partial-sum scratch
  int bh = blockIdx.y;                      // 0..31
  int h = bh % HHD, b = bh / HHD;
  int rowBase = blockIdx.x * 16;            // query rows within N
  int tid = threadIdx.x, wave = tid >> 5, lane = tid & 31;
  int m = lane & 15, hf = lane >> 4;

  // A fragments (Q rows, K=64 -> two 16x32 chunks)
  Frag16 aq[2];
  const u16* qrow = qb + ((size_t)bh * NN + rowBase + m) * HD;
  loadA(aq[0], qrow, hf);
  loadA(aq[1], qrow + 32, hf);

  const float scale = 0.125f;               // 1/sqrt(64)
  const float ascale = adj_scale[0];

  // ---- phase 1: scores -> LDS (8 waves x 8 col tiles = 1024 cols)
  for (int t = 0; t < 8; ++t) {
    int ncol = (wave * 8 + t) * 16 + m;
    const u16* kcol = kb + ((size_t)bh * NN + ncol) * HD;
    Frag16 bk0, bk1;
    loadB(bk0, kcol, hf);                   // both K chunks in flight
    loadB(bk1, kcol + 32, hf);
    v8f acc = {};
    acc = wmma_bf16(aq[0], bk0, acc);
    acc = wmma_bf16(aq[1], bk1, acc);
#pragma unroll
    for (int j = 0; j < 8; ++j) {
      int mm = j + hf * 8;
      int grow = rowBase + mm;              // query index in N
      float s = acc[j] * scale + attn_bias[((size_t)h * NN + grow) * NN + ncol];
      s += ascale * ((adj[(size_t)grow * NN + ncol] > 0.f) ? 0.f : -1e9f);
      Sc[mm * NN + ncol] = s;
    }
  }
  __syncthreads();

  // ---- phase 2: per-row kth-largest (radix bit-select) + softmax, one wave per row
  float th = 1.f / (1.f + __expf(-sth[h]));
  int kval = (int)(NN * (1.f - th)); if (kval < 1) kval = 1;
  for (int rr = 0; rr < 2; ++rr) {
    int r = wave * 2 + rr;
    float vals[32]; u32 keys[32];
    float mx = -3.4e38f;
#pragma unroll
    for (int i = 0; i < 32; ++i) {
      float v = Sc[r * NN + i * 32 + lane];
      vals[i] = v;
      u32 u = __float_as_uint(v);
      keys[i] = (u & 0x80000000u) ? ~u : (u | 0x80000000u);   // order-preserving
      mx = fmaxf(mx, v);
    }
#pragma unroll
    for (int s = 16; s > 0; s >>= 1) mx = fmaxf(mx, __shfl_xor(mx, s, 32));
    u32 prefix = 0;
    for (int bit = 31; bit >= 0; --bit) {   // 32-step MSB-first select
      u32 cand = prefix | (1u << bit);
      int cnt = 0;
#pragma unroll
      for (int i = 0; i < 32; ++i) cnt += (keys[i] >= cand);
#pragma unroll
      for (int s = 16; s > 0; s >>= 1) cnt += __shfl_xor(cnt, s, 32);
      if (cnt >= kval) prefix = cand;
    }
    float psum = 0.f; float p[32];
#pragma unroll
    for (int i = 0; i < 32; ++i) {
      float e = (keys[i] >= prefix) ? __expf(vals[i] - mx) : 0.f;
      p[i] = e; psum += e;
    }
#pragma unroll
    for (int s = 16; s > 0; s >>= 1) psum += __shfl_xor(psum, s, 32);
    float inv = 1.f / psum;
    u16* arow = (u16*)&Sc[r * NN];          // bf16 attn aliases own row (regs hold data)
#pragma unroll
    for (int i = 0; i < 32; ++i) arow[i * 32 + lane] = f2bf(p[i] * inv);
  }
  __syncthreads();

  // ---- phase 3: attended = attn @ V, all 8 waves: dtile = wave&3, K-half = wave>>2
  {
    int dtile = wave & 3, khalf = wave >> 2;
    int d = dtile * 16 + m;
    const u16* abase = (const u16*)Sc + m * 2048 + khalf * 512;  // attn row, 2048-u16 stride
    const u16* vbase = vT + ((size_t)bh * HD + d) * NN + khalf * 512;
    v8f acc = {};
    Frag16 a0, a1, b0, b1;
    loadA(a0, abase, hf);
    loadB(b0, vbase, hf);
#pragma unroll
    for (int kc = 0; kc < 16; ++kc) {
      Frag16& ac = (kc & 1) ? a1 : a0;
      Frag16& bc = (kc & 1) ? b1 : b0;
      Frag16& an = (kc & 1) ? a0 : a1;
      Frag16& bn = (kc & 1) ? b0 : b1;
      if (kc + 1 < 16) {                    // pipeline next chunk
        loadA(an, abase + (kc + 1) * 32, hf);
        loadB(bn, vbase + (kc + 1) * 32, hf);
      }
      acc = wmma_bf16(ac, bc, acc);
    }
    if (wave >= 4) {
#pragma unroll
      for (int j = 0; j < 8; ++j) Red[dtile * 256 + j * 32 + lane] = acc[j];
    }
    __syncthreads();
    if (wave < 4) {
#pragma unroll
      for (int j = 0; j < 8; ++j) {
        float v = acc[j] + Red[dtile * 256 + j * 32 + lane];
        int grow = rowBase + j + hf * 8;
        attended[((size_t)b * NN + grow) * CCH + h * HD + d] = f2bf(v);
      }
    }
  }
}

// ---------------------------------------------------------------- out proj + residual
__global__ __launch_bounds__(256) void out_gemm(const u16* __restrict__ att,
                                                const u16* __restrict__ wout,
                                                const float* __restrict__ out_b,
                                                const float* __restrict__ x,
                                                float* __restrict__ out) {
  __shared__ u16 As[16 * CCH];
  int rowBase = blockIdx.x * 16;
  int tid = threadIdx.x;
  stage_tile(As, att + (size_t)rowBase * CCH, tid);
  __syncthreads();
  int wave = tid >> 5, lane = tid & 31;
  int m = lane & 15, hf = lane >> 4;
  for (int it = 0; it < 4; ++it) {
    int oc = (wave + it * 8) * 16 + m;      // col tiles 0..31
    const u16* wbase = wout + (size_t)oc * CCH;
    v8f acc = {};
    Frag16 bw0, bw1;
    loadB(bw0, wbase, hf);
#pragma unroll
    for (int kc = 0; kc < 16; ++kc) {
      Frag16& cur = (kc & 1) ? bw1 : bw0;
      Frag16& nxt = (kc & 1) ? bw0 : bw1;
      if (kc + 1 < 16) loadB(nxt, wbase + (kc + 1) * 32, hf);
      Frag16 a;
      loadA(a, &As[m * CCH + kc * 32], hf);
      acc = wmma_bf16(a, cur, acc);
    }
    float bias = out_b[oc];
#pragma unroll
    for (int j = 0; j < 8; ++j) {
      int row = rowBase + j + hf * 8;
      size_t idx = (size_t)row * CCH + oc;
      out[idx] = acc[j] + bias + x[idx];    // residual
    }
  }
}

// ---------------------------------------------------------------- reg loss
// softmax rows each sum to 1 (kval>=1) => mean|attn| == 1/N exactly.
__global__ void reg_kernel(const float* __restrict__ w, float* __restrict__ out) {
  if (threadIdx.x == 0) out[0] = log1pf(__expf(w[0])) * (1.0f / NN);
}

extern "C" void kernel_launch(void* const* d_in, const int* in_sizes, int n_in,
                              void* d_out, int out_size, void* d_ws, size_t ws_size,
                              hipStream_t stream) {
  const float* x         = (const float*)d_in[0];
  const float* adj       = (const float*)d_in[1];
  const float* ln_g      = (const float*)d_in[2];
  const float* ln_b      = (const float*)d_in[3];
  const float* qkv_w     = (const float*)d_in[4];
  const float* qkv_b     = (const float*)d_in[5];
  const float* out_w     = (const float*)d_in[6];
  const float* out_b     = (const float*)d_in[7];
  const float* attn_bias = (const float*)d_in[8];
  const float* adj_scale = (const float*)d_in[9];
  const float* sth       = (const float*)d_in[10];
  const float* l1w       = (const float*)d_in[11];
  float* out = (float*)d_out;

  // bf16 workspace carve-up (~23 MB)
  u16* wqkv = (u16*)d_ws;                   // 3C*C
  u16* wout = wqkv + (size_t)C3 * CCH;      // C*C
  u16* xn   = wout + (size_t)CCH * CCH;     // BN*C
  u16* qb   = xn   + (size_t)BN * CCH;      // [bh,n,d]
  u16* kb   = qb   + (size_t)BN * CCH;      // [bh,n,d]
  u16* vT   = kb   + (size_t)BN * CCH;      // [bh,d,n]
  u16* att  = vT   + (size_t)BN * CCH;      // [BN,C]

  cvt_weights<<<(C3 * CCH + 255) / 256, 256, 0, stream>>>(qkv_w, out_w, wqkv, wout);
  ln_kernel<<<BN / 8, 256, 0, stream>>>(x, ln_g, ln_b, xn);
  qkv_gemm<<<BN / 16, 256, 0, stream>>>(xn, wqkv, qkv_b, qb, kb, vT);
  attn_kernel<<<dim3(NN / 16, BB * HHD), 256, 0, stream>>>(qb, kb, vT, attn_bias, adj,
                                                           adj_scale, sth, att);
  out_gemm<<<BN / 16, 256, 0, stream>>>(att, wout, out_b, x, out);
  reg_kernel<<<1, 32, 0, stream>>>(l1w, out + (size_t)BN * CCH);
}
